// ComplexEdgeMessages_6631429505706
// MI455X (gfx1250) — matrix operand order repaired and checked by reading, not compiled, not run on hardware
//
#include <hip/hip_runtime.h>
#include <hip/hip_bf16.h>

typedef __attribute__((ext_vector_type(16))) _Float16 v16h;
typedef __attribute__((ext_vector_type(8)))  float    v8f;

union AOp { _Float16 h[16]; unsigned u[8]; v16h v; };
union Acc { float f[8]; v8f v; };

#define EPS_LN 1e-5f

__device__ __forceinline__ float silu_f(float x) {
    return x / (1.f + __expf(-x));
}

// ---------------------------------------------------------------------------
// Stage 1a: per-ligand-node: LN(384) -> ll_proj(128), lig_p(64)
//           -> a_ll = W1s@proj, b_ll = W1e@proj, s_pl = P1s@lig_p
// ---------------------------------------------------------------------------
__global__ __launch_bounds__(128) void lig_row_kernel(
    const float* __restrict__ feats,      // (1024, 384)
    const float* __restrict__ g, const float* __restrict__ bv,  // (384)
    const float* __restrict__ W_lig, const float* __restrict__ b_lig,       // (128,384),(128)
    const float* __restrict__ W_ligpro, const float* __restrict__ b_ligpro, // (64,384),(64)
    const float* __restrict__ llW1,       // (128, 384)
    const float* __restrict__ plW1,       // (64, 192)
    float* __restrict__ a_ll, float* __restrict__ b_ll, float* __restrict__ s_pl)
{
    __shared__ float row[384];
    __shared__ float red[128];
    __shared__ float proj[128];
    __shared__ float ligp[64];
    const int t = threadIdx.x;
    const int bi = blockIdx.x;
    const float* x = feats + (size_t)bi * 384;

    float v0 = x[t], v1 = x[t + 128], v2 = x[t + 256];
    red[t] = v0 + v1 + v2;
    __syncthreads();
    for (int off = 64; off > 0; off >>= 1) { if (t < off) red[t] += red[t + off]; __syncthreads(); }
    float mean = red[0] * (1.f / 384.f);
    __syncthreads();
    red[t] = v0 * v0 + v1 * v1 + v2 * v2;
    __syncthreads();
    for (int off = 64; off > 0; off >>= 1) { if (t < off) red[t] += red[t + off]; __syncthreads(); }
    float var = red[0] * (1.f / 384.f) - mean * mean;
    float inv = rsqrtf(var + EPS_LN);
    for (int k = t; k < 384; k += 128) row[k] = (x[k] - mean) * inv * g[k] + bv[k];
    __syncthreads();

    {   // ll_proj row
        float acc = b_lig[t];
        #pragma unroll 4
        for (int k = 0; k < 384; ++k) acc += W_lig[(size_t)t * 384 + k] * row[k];
        proj[t] = acc;
    }
    if (t < 64) {
        float acc = b_ligpro[t];
        #pragma unroll 4
        for (int k = 0; k < 384; ++k) acc += W_ligpro[(size_t)t * 384 + k] * row[k];
        ligp[t] = acc;
    }
    __syncthreads();
    {
        float a = 0.f, b = 0.f;
        #pragma unroll 4
        for (int k = 0; k < 128; ++k) {
            a += llW1[(size_t)t * 384 + k] * proj[k];
            b += llW1[(size_t)t * 384 + 128 + k] * proj[k];
        }
        a_ll[(size_t)bi * 128 + t] = a;
        b_ll[(size_t)bi * 128 + t] = b;
    }
    if (t < 64) {
        float acc = 0.f;
        #pragma unroll 4
        for (int k = 0; k < 64; ++k) acc += plW1[(size_t)t * 192 + k] * ligp[k];
        s_pl[(size_t)bi * 64 + t] = acc;
    }
}

// ---------------------------------------------------------------------------
// Stage 1b: per-protein-node: LN(512) -> pro_p(64) -> e_pl = P1e@pro_p
// ---------------------------------------------------------------------------
__global__ __launch_bounds__(128) void pro_row_kernel(
    const float* __restrict__ feats,      // (3072, 512)
    const float* __restrict__ g, const float* __restrict__ bv,  // (512)
    const float* __restrict__ W_pro, const float* __restrict__ b_pro, // (64,512),(64)
    const float* __restrict__ plW1,       // (64, 192)
    float* __restrict__ e_pl)
{
    __shared__ float row[512];
    __shared__ float red[128];
    __shared__ float prop[64];
    const int t = threadIdx.x;
    const int bi = blockIdx.x;
    const float* x = feats + (size_t)bi * 512;

    float s0 = 0.f, s1 = 0.f;
    for (int k = t; k < 512; k += 128) { float v = x[k]; s0 += v; s1 += v * v; }
    red[t] = s0;
    __syncthreads();
    for (int off = 64; off > 0; off >>= 1) { if (t < off) red[t] += red[t + off]; __syncthreads(); }
    float mean = red[0] * (1.f / 512.f);
    __syncthreads();
    red[t] = s1;
    __syncthreads();
    for (int off = 64; off > 0; off >>= 1) { if (t < off) red[t] += red[t + off]; __syncthreads(); }
    float var = red[0] * (1.f / 512.f) - mean * mean;
    float inv = rsqrtf(var + EPS_LN);
    for (int k = t; k < 512; k += 128) row[k] = (x[k] - mean) * inv * g[k] + bv[k];
    __syncthreads();
    if (t < 64) {
        float acc = b_pro[t];
        #pragma unroll 4
        for (int k = 0; k < 512; ++k) acc += W_pro[(size_t)t * 512 + k] * row[k];
        prop[t] = acc;
    }
    __syncthreads();
    if (t < 64) {
        float acc = 0.f;
        #pragma unroll 4
        for (int k = 0; k < 64; ++k) acc += plW1[(size_t)t * 192 + 64 + k] * prop[k];
        e_pl[(size_t)bi * 64 + t] = acc;
    }
}

// ---------------------------------------------------------------------------
// Stage 2a: ligand-ligand pair kernel.  1 workgroup = (b,i); 8 waves, each
// wave owns one 16-j tile (M = j).  GEMM1 K=64 dots + K=64 edge, GEMM2 K=128.
// ---------------------------------------------------------------------------
__global__ __launch_bounds__(256) void ll_pair_kernel(
    const float* __restrict__ lig_coords,   // (8,64,128,3)
    const float* __restrict__ edge,         // (8,128,128,64)
    const float* __restrict__ eg, const float* __restrict__ eb, // (64)
    const float* __restrict__ llW1,         // (128,384)
    const float* __restrict__ llb1,         // (128)
    const float* __restrict__ llW2,         // (128,128)
    const float* __restrict__ llb2,         // (128)
    const float* __restrict__ a_ll,         // (1024,128)
    const float* __restrict__ b_ll,         // (1024,128)
    float* __restrict__ out)                // (8,128,512,128)
{
    __shared__ _Float16 s_w1c[128 * 64];    // [f][s]  (k-contiguous for B tiles)
    __shared__ _Float16 s_w1d[128 * 64];    // [f][d]
    __shared__ _Float16 s_w2 [128 * 128];   // [o][f]
    __shared__ float    s_arow[128];
    __shared__ float    s_b1[128], s_b2[128];
    __shared__ float    s_eg[64], s_eb[64];
    __shared__ _Float16 s_silu[8][16 * 128]; // per-wave silu(h) staging

    const int tid = threadIdx.x;
    const int bi  = blockIdx.x;          // b*128 + i
    const int b   = bi >> 7;
    const int i   = bi & 127;

    for (int idx = tid; idx < 128 * 64; idx += 256) {
        int f = idx >> 6, s = idx & 63;
        s_w1c[idx] = (_Float16)llW1[(size_t)f * 384 + 256 + s];
        s_w1d[idx] = (_Float16)llW1[(size_t)f * 384 + 320 + s];
    }
    for (int idx = tid; idx < 128 * 128; idx += 256)
        s_w2[idx] = (_Float16)llW2[idx];
    if (tid < 128) {
        s_arow[tid] = a_ll[(size_t)bi * 128 + tid];
        s_b1[tid] = llb1[tid];
        s_b2[tid] = llb2[tid];
    }
    if (tid < 64) { s_eg[tid] = eg[tid]; s_eb[tid] = eb[tid]; }
    __syncthreads();

    const int wave = tid >> 5;
    const int lane = tid & 31;
    const int m    = lane & 15;          // M row within tile (== j offset, == N col for B/C)
    const int half = lane >> 4;
    const int j0   = wave * 16;
    const int j    = j0 + m;
    _Float16* silu_buf = &s_silu[wave][0];

    // ---- A operands: coordinate dots (16 pairs x 64 s), two 16x32 tiles ----
    const float* ci = lig_coords + (size_t)b * 64 * 128 * 3 + (size_t)i * 3;
    const float* cj = lig_coords + (size_t)b * 64 * 128 * 3 + (size_t)j * 3;
    AOp Ad[2];
    #pragma unroll
    for (int t = 0; t < 2; ++t)
        #pragma unroll
        for (int v = 0; v < 8; ++v) {
            const int k0 = t * 32 + ((v < 4) ? 0 : 16) + half * 8 + (v & 3) * 2;
            #pragma unroll
            for (int e = 0; e < 2; ++e) {
                const int s = k0 + e;
                const float* pi = ci + (size_t)s * 128 * 3;
                const float* pj = cj + (size_t)s * 128 * 3;
                Ad[t].h[v * 2 + e] =
                    (_Float16)(pi[0] * pj[0] + pi[1] * pj[1] + pi[2] * pj[2]);
            }
        }

    // ---- A operands: edge features, LayerNorm(64) on the fly ----
    const float* erow = edge + ((((size_t)(b * 128 + i)) * 128) + j) * 64;
    float vals[32];
    float sum = 0.f, sq = 0.f;
    {
        int kk = 0;
        #pragma unroll
        for (int t = 0; t < 2; ++t)
            #pragma unroll
            for (int v = 0; v < 8; ++v) {
                const int k0 = t * 32 + ((v < 4) ? 0 : 16) + half * 8 + (v & 3) * 2;
                float e0 = erow[k0], e1 = erow[k0 + 1];
                vals[kk++] = e0; vals[kk++] = e1;
                sum += e0 + e1; sq += e0 * e0 + e1 * e1;
            }
    }
    sum += __shfl_xor(sum, 16, 32);      // lanes L and L^16 share pair m
    sq  += __shfl_xor(sq, 16, 32);
    const float mean = sum * (1.f / 64.f);
    const float inv  = rsqrtf(sq * (1.f / 64.f) - mean * mean + EPS_LN);
    AOp Ae[2];
    {
        int kk = 0;
        #pragma unroll
        for (int t = 0; t < 2; ++t)
            #pragma unroll
            for (int v = 0; v < 8; ++v) {
                const int k0 = t * 32 + ((v < 4) ? 0 : 16) + half * 8 + (v & 3) * 2;
                Ae[t].h[v * 2 + 0] = (_Float16)((vals[kk] - mean) * inv * s_eg[k0] + s_eb[k0]);
                Ae[t].h[v * 2 + 1] = (_Float16)((vals[kk + 1] - mean) * inv * s_eg[k0 + 1] + s_eb[k0 + 1]);
                kk += 2;
            }
    }

    // ---- GEMM1: h tiles over f (8 tiles of 16), silu -> LDS ----
    #pragma unroll
    for (int ft = 0; ft < 8; ++ft) {
        Acc acc = {};
        const int nb = ft * 16 + m;      // B/C column = f index
        #pragma unroll
        for (int kt = 0; kt < 2; ++kt) {
            AOp Bc, Bd;
            const int kb = kt * 32 + half * 8;
            #pragma unroll
            for (int v = 0; v < 8; ++v) {
                Bc.u[v] = *(const unsigned*)&s_w1c[nb * 64 + kb + ((v < 4) ? 0 : 16) + (v & 3) * 2];
                Bd.u[v] = *(const unsigned*)&s_w1d[nb * 64 + kb + ((v < 4) ? 0 : 16) + (v & 3) * 2];
            }
            acc.v = __builtin_amdgcn_wmma_f32_16x16x32_f16(false, Ad[kt].v, false, Bc.v,
                                                           (short)0, acc.v, false, false);
            acc.v = __builtin_amdgcn_wmma_f32_16x16x32_f16(false, Ae[kt].v, false, Bd.v,
                                                           (short)0, acc.v, false, false);
        }
        const float af = s_arow[nb] + s_b1[nb];
        #pragma unroll
        for (int r = 0; r < 8; ++r) {
            const int mm = half * 8 + r;
            float hv = acc.f[r] + af + b_ll[((size_t)(b * 128 + j0 + mm)) * 128 + nb];
            silu_buf[mm * 128 + nb] = (_Float16)silu_f(hv);
        }
    }

    // ---- GEMM2: silu(h)(16x128) @ W2^T, K=128, N=128 ----
    AOp A2[4];
    #pragma unroll
    for (int kt = 0; kt < 4; ++kt)
        #pragma unroll
        for (int v = 0; v < 8; ++v) {
            const int k0 = kt * 32 + ((v < 4) ? 0 : 16) + half * 8 + (v & 3) * 2;
            A2[kt].u[v] = *(const unsigned*)&silu_buf[m * 128 + k0];
        }
    float* outp = out + (((size_t)(b * 128 + i)) * 512 + j0) * 128;
    #pragma unroll
    for (int ot = 0; ot < 8; ++ot) {
        Acc acc = {};
        const int nb = ot * 16 + m;
        #pragma unroll
        for (int kt = 0; kt < 4; ++kt) {
            AOp Bw;
            const int kb = kt * 32 + half * 8;
            #pragma unroll
            for (int v = 0; v < 8; ++v)
                Bw.u[v] = *(const unsigned*)&s_w2[nb * 128 + kb + ((v < 4) ? 0 : 16) + (v & 3) * 2];
            acc.v = __builtin_amdgcn_wmma_f32_16x16x32_f16(false, A2[kt].v, false, Bw.v,
                                                           (short)0, acc.v, false, false);
        }
        const float b2v = s_b2[nb];
        #pragma unroll
        for (int r = 0; r < 8; ++r)
            outp[(size_t)(half * 8 + r) * 128 + nb] = acc.f[r] + b2v;
    }
}

// ---------------------------------------------------------------------------
// Stage 2b: ligand-protein pair kernel.  1 workgroup = (b,i); 8 waves, each
// wave loops over 3 of the 24 j-tiles.  GEMM1 N=64 K=64, GEMM2 N=128 K=64.
// ---------------------------------------------------------------------------
__global__ __launch_bounds__(256) void pl_pair_kernel(
    const float* __restrict__ lig_coords,   // (8,64,128,3)
    const float* __restrict__ pro_coords,   // (8,64,384,3)
    const float* __restrict__ plW1,         // (64,192)
    const float* __restrict__ plb1,         // (64)
    const float* __restrict__ plW2,         // (128,64)
    const float* __restrict__ plb2,         // (128)
    const float* __restrict__ s_pl,         // (1024,64)
    const float* __restrict__ e_pl,         // (3072,64)
    float* __restrict__ out)                // (8,128,512,128)
{
    __shared__ _Float16 s_p1c[64 * 64];     // [f][s]
    __shared__ _Float16 s_pw2[128 * 64];    // [o][f]
    __shared__ float    s_srow[64];
    __shared__ float    s_b1[64], s_b2[128];
    __shared__ _Float16 s_silu[8][16 * 64];

    const int tid = threadIdx.x;
    const int bi  = blockIdx.x;
    const int b   = bi >> 7;
    const int i   = bi & 127;

    for (int idx = tid; idx < 64 * 64; idx += 256) {
        int f = idx >> 6, s = idx & 63;
        s_p1c[idx] = (_Float16)plW1[(size_t)f * 192 + 128 + s];
    }
    for (int idx = tid; idx < 128 * 64; idx += 256)
        s_pw2[idx] = (_Float16)plW2[idx];
    if (tid < 64) { s_srow[tid] = s_pl[(size_t)bi * 64 + tid]; s_b1[tid] = plb1[tid]; }
    if (tid < 128) s_b2[tid] = plb2[tid];
    __syncthreads();

    const int wave = tid >> 5;
    const int lane = tid & 31;
    const int m    = lane & 15;
    const int half = lane >> 4;
    _Float16* silu_buf = &s_silu[wave][0];

    const float* ci = lig_coords + (size_t)b * 64 * 128 * 3 + (size_t)i * 3;

    for (int jt = wave; jt < 24; jt += 8) {
        const int j0 = jt * 16;
        const int j  = j0 + m;

        // ---- A: lig-pro coordinate dots ----
        AOp Ad[2];
        const float* cj = pro_coords + (size_t)b * 64 * 384 * 3 + (size_t)j * 3;
        #pragma unroll
        for (int t = 0; t < 2; ++t)
            #pragma unroll
            for (int v = 0; v < 8; ++v) {
                const int k0 = t * 32 + ((v < 4) ? 0 : 16) + half * 8 + (v & 3) * 2;
                #pragma unroll
                for (int e = 0; e < 2; ++e) {
                    const int s = k0 + e;
                    const float* pi = ci + (size_t)s * 128 * 3;
                    const float* pj = cj + (size_t)s * 384 * 3;
                    Ad[t].h[v * 2 + e] =
                        (_Float16)(pi[0] * pj[0] + pi[1] * pj[1] + pi[2] * pj[2]);
                }
            }

        // ---- GEMM1: 4 f-tiles of 16 (d_ffp = 64) ----
        #pragma unroll
        for (int ft = 0; ft < 4; ++ft) {
            Acc acc = {};
            const int nb = ft * 16 + m;
            #pragma unroll
            for (int kt = 0; kt < 2; ++kt) {
                AOp Bc;
                const int kb = kt * 32 + half * 8;
                #pragma unroll
                for (int v = 0; v < 8; ++v)
                    Bc.u[v] = *(const unsigned*)&s_p1c[nb * 64 + kb + ((v < 4) ? 0 : 16) + (v & 3) * 2];
                acc.v = __builtin_amdgcn_wmma_f32_16x16x32_f16(false, Ad[kt].v, false, Bc.v,
                                                               (short)0, acc.v, false, false);
            }
            const float sf = s_srow[nb] + s_b1[nb];
            #pragma unroll
            for (int r = 0; r < 8; ++r) {
                const int mm = half * 8 + r;
                float hv = acc.f[r] + sf + e_pl[((size_t)(b * 384 + j0 + mm)) * 64 + nb];
                silu_buf[mm * 64 + nb] = (_Float16)silu_f(hv);
            }
        }

        // ---- GEMM2: silu(h2)(16x64) @ pl_W2^T, N=128, K=64 ----
        AOp A2[2];
        #pragma unroll
        for (int kt = 0; kt < 2; ++kt)
            #pragma unroll
            for (int v = 0; v < 8; ++v) {
                const int k0 = kt * 32 + ((v < 4) ? 0 : 16) + half * 8 + (v & 3) * 2;
                A2[kt].u[v] = *(const unsigned*)&silu_buf[m * 64 + k0];
            }
        float* outp = out + (((size_t)(b * 128 + i)) * 512 + 128 + j0) * 128;
        #pragma unroll
        for (int ot = 0; ot < 8; ++ot) {
            Acc acc = {};
            const int nb = ot * 16 + m;
            #pragma unroll
            for (int kt = 0; kt < 2; ++kt) {
                AOp Bw;
                const int kb = kt * 32 + half * 8;
                #pragma unroll
                for (int v = 0; v < 8; ++v)
                    Bw.u[v] = *(const unsigned*)&s_pw2[nb * 64 + kb + ((v < 4) ? 0 : 16) + (v & 3) * 2];
                acc.v = __builtin_amdgcn_wmma_f32_16x16x32_f16(false, A2[kt].v, false, Bw.v,
                                                               (short)0, acc.v, false, false);
            }
            const float b2v = s_b2[nb];
            #pragma unroll
            for (int r = 0; r < 8; ++r)
                outp[(size_t)(half * 8 + r) * 128 + nb] = acc.f[r] + b2v;
        }
    }
}

// ---------------------------------------------------------------------------
extern "C" void kernel_launch(void* const* d_in, const int* in_sizes, int n_in,
                              void* d_out, int out_size, void* d_ws, size_t ws_size,
                              hipStream_t stream) {
    const float* lig_coords = (const float*)d_in[0];
    const float* lig_feats  = (const float*)d_in[1];
    // d_in[2]: lig_node_mask (unused by reference)
    const float* lig_edge   = (const float*)d_in[3];
    const float* pro_coords = (const float*)d_in[4];
    const float* pro_feats  = (const float*)d_in[5];
    // d_in[6]: pro_node_mask (unused)
    const float* ln_lig_g   = (const float*)d_in[7];
    const float* ln_lig_b   = (const float*)d_in[8];
    const float* ln_pro_g   = (const float*)d_in[9];
    const float* ln_pro_b   = (const float*)d_in[10];
    const float* ln_edge_g  = (const float*)d_in[11];
    const float* ln_edge_b  = (const float*)d_in[12];
    const float* W_lig      = (const float*)d_in[13];
    const float* b_lig      = (const float*)d_in[14];
    const float* W_ligpro   = (const float*)d_in[15];
    const float* b_ligpro   = (const float*)d_in[16];
    const float* W_pro      = (const float*)d_in[17];
    const float* b_pro      = (const float*)d_in[18];
    const float* llW1       = (const float*)d_in[19];
    const float* llb1       = (const float*)d_in[20];
    const float* llW2       = (const float*)d_in[21];
    const float* llb2       = (const float*)d_in[22];
    const float* plW1       = (const float*)d_in[23];
    const float* plb1       = (const float*)d_in[24];
    const float* plW2       = (const float*)d_in[25];
    const float* plb2       = (const float*)d_in[26];
    float* out = (float*)d_out;

    // workspace layout (floats): a_ll | b_ll | s_pl | e_pl  (2 MB total)
    float* ws   = (float*)d_ws;
    float* a_ll = ws;
    float* b_ll = ws + 131072;
    float* s_pl = ws + 262144;
    float* e_pl = ws + 327680;

    lig_row_kernel<<<1024, 128, 0, stream>>>(lig_feats, ln_lig_g, ln_lig_b,
                                             W_lig, b_lig, W_ligpro, b_ligpro,
                                             llW1, plW1, a_ll, b_ll, s_pl);
    pro_row_kernel<<<3072, 128, 0, stream>>>(pro_feats, ln_pro_g, ln_pro_b,
                                             W_pro, b_pro, plW1, e_pl);
    ll_pair_kernel<<<1024, 256, 0, stream>>>(lig_coords, lig_edge, ln_edge_g, ln_edge_b,
                                             llW1, llb1, llW2, llb2, a_ll, b_ll, out);
    pl_pair_kernel<<<1024, 256, 0, stream>>>(lig_coords, pro_coords,
                                             plW1, plb1, plW2, plb2, s_pl, e_pl, out);
}
